// SimpleTransformer_28063316312178
// MI455X (gfx1250) — matrix operand order repaired and checked by reading, not compiled, not run on hardware
//
#include <hip/hip_runtime.h>
#include <hip/hip_bf16.h>
#include <math.h>

#define NHEADS 8
#define DMODEL 512
#define DK     64
#define SEQL   2048
#define NBATCH 2
#define NROWS  (NBATCH*SEQL)   // 4096
#define NVOCAB 32000

typedef __attribute__((ext_vector_type(16))) __bf16 v16bf;
typedef __attribute__((ext_vector_type(8)))  __bf16 v8bf;
typedef __attribute__((ext_vector_type(8)))  float  v8f;

__device__ __forceinline__ v8f wmma_bf16(v16bf a, v16bf b, v8f c) {
  return __builtin_amdgcn_wmma_f32_16x16x32_bf16(false, a, false, b, (short)0, c, false, false);
}
__device__ __forceinline__ v16bf cat8(v8bf lo, v8bf hi) {
  return __builtin_shufflevector(lo, hi, 0,1,2,3,4,5,6,7,8,9,10,11,12,13,14,15);
}

// ---------------- fp32 -> bf16 convert ----------------
__global__ void k_cvt_bf16(const float* __restrict__ x, __bf16* __restrict__ y, int n) {
  int i = blockIdx.x*blockDim.x + threadIdx.x;
  if (i < n) y[i] = (__bf16)x[i];
}

// ---------------- GEMM: C[M,N] = A[M,K](bf16) @ W[N,K]^T(bf16) + bias, opt ReLU ----
// block = 256 thr = 8 waves; wave -> 16x64 strip; block -> 128x64 tile
__global__ __launch_bounds__(256)
void k_gemm_bf16(const __bf16* __restrict__ A, const __bf16* __restrict__ W,
                 const float* __restrict__ bias,
                 float* __restrict__ Cf, __bf16* __restrict__ Cb,
                 int M, int N, int K, int relu)
{
  const int lane  = threadIdx.x & 31;
  const int wave  = threadIdx.x >> 5;
  const int mbase = blockIdx.x*128 + wave*16;
  const int nbase = blockIdx.y*64;
  const int mrow  = lane & 15;     // A row / B col / C col within tile
  const int half  = lane >> 4;

  const __bf16* ap = A + (size_t)(mbase + mrow)*K;
  v8f acc[4] = {};
  for (int kb = 0; kb < K; kb += 32) {
    v16bf av = cat8(*(const v8bf*)(ap + kb + 16*half),
                    *(const v8bf*)(ap + kb + 16*half + 8));
    #pragma unroll
    for (int j = 0; j < 4; ++j) {
      const __bf16* wp = W + (size_t)(nbase + 16*j + mrow)*K + kb + 16*half;
      acc[j] = wmma_bf16(av, *(const v16bf*)wp, acc[j]);
    }
  }
  #pragma unroll
  for (int j = 0; j < 4; ++j) {
    const int col = nbase + 16*j + mrow;
    const float bv = bias ? bias[col] : 0.f;
    #pragma unroll
    for (int r = 0; r < 8; ++r) {
      const size_t row = (size_t)(mbase + r + 8*half);
      float v = acc[j][r] + bv;
      if (relu) v = fmaxf(v, 0.f);
      if (Cf) Cf[row*(size_t)N + col] = v;
      if (Cb) Cb[row*(size_t)N + col] = (__bf16)v;
    }
  }
}

// ---------------- split heads, cvt to bf16, optional transpose ----------------
// in : f32 [B*L, D]   out(!tr): bf16 [B*NH, L, DK]   out(tr): bf16 [B*NH, DK, L]
__global__ void k_pack_heads(const float* __restrict__ x, __bf16* __restrict__ y, int tr)
{
  int i = blockIdx.x*blockDim.x + threadIdx.x;
  if (i >= NROWS*DMODEL) return;
  int row = i / DMODEL, col = i % DMODEL;
  int b = row / SEQL, t = row % SEQL;
  int h = col / DK,   d = col % DK;
  size_t o = tr ? (((size_t)(b*NHEADS + h)*DK + d)*SEQL + t)
                : (((size_t)(b*NHEADS + h)*SEQL + t)*DK + d);
  y[o] = (__bf16)x[i];
}

// ---------------- wave-level flash attention ----------------
// Q,K: bf16 [B*NH, L, DK]   Vt: bf16 [B*NH, DK, L]   O: bf16 concat [B*L, D]
__global__ __launch_bounds__(256)
void k_flash_attn(const __bf16* __restrict__ Q, const __bf16* __restrict__ K,
                  const __bf16* __restrict__ Vt, __bf16* __restrict__ O, int causal)
{
  __shared__ __bf16 plds[8][16*32];       // 1KB per wave P bounce
  const int lane  = threadIdx.x & 31;
  const int wave  = threadIdx.x >> 5;
  const int bh    = blockIdx.x;
  const int qbase = blockIdx.y*128 + wave*16;
  const int b = bh / NHEADS, h = bh % NHEADS;
  const int mrow = lane & 15;
  const int half = lane >> 4;

  const __bf16* qp = Q + ((size_t)bh*SEQL + qbase + mrow)*DK;
  const v16bf qa0 = cat8(*(const v8bf*)(qp + 16*half), *(const v8bf*)(qp + 16*half + 8));
  const v16bf qa1 = cat8(*(const v8bf*)(qp + 32 + 16*half), *(const v8bf*)(qp + 40 + 16*half));

  v8f o[4] = {};
  float mrun[8], lrun[8];
  #pragma unroll
  for (int r = 0; r < 8; ++r) { mrun[r] = -INFINITY; lrun[r] = 0.f; }

  const int nblk = causal ? (qbase/32 + 1) : (SEQL/32);
  for (int blk = 0; blk < nblk; ++blk) {
    const int kbase = blk*32;
    const bool maskblk = (causal && blk == nblk-1);

    // S = Q K^T over dk=64 (two k-steps per 16-key tile)
    const __bf16* kp = K + ((size_t)bh*SEQL + kbase + mrow)*DK + 16*half;
    v8f s0 = {}, s1 = {};
    s0 = wmma_bf16(qa0, *(const v16bf*)(kp),           s0);
    s0 = wmma_bf16(qa1, *(const v16bf*)(kp + 32),      s0);
    s1 = wmma_bf16(qa0, *(const v16bf*)(kp + 16*DK),   s1);
    s1 = wmma_bf16(qa1, *(const v16bf*)(kp + 16*DK+32),s1);

    float p0[8], p1[8];
    #pragma unroll
    for (int r = 0; r < 8; ++r) {
      const int qi = qbase + r + 8*half;
      float a = s0[r]*0.125f;          // 1/sqrt(64)
      float c = s1[r]*0.125f;
      if (maskblk) {
        if (kbase + mrow      > qi) a = -INFINITY;
        if (kbase + 16 + mrow > qi) c = -INFINITY;
      }
      float rmax = fmaxf(a, c);
      #pragma unroll
      for (int m = 1; m < 16; m <<= 1) rmax = fmaxf(rmax, __shfl_xor(rmax, m, 32));
      const float mnew = fmaxf(mrun[r], rmax);
      const float e0 = __expf(a - mnew);
      const float e1 = __expf(c - mnew);
      float rs = e0 + e1;
      #pragma unroll
      for (int m = 1; m < 16; m <<= 1) rs += __shfl_xor(rs, m, 32);
      const float corr = __expf(mrun[r] - mnew);
      lrun[r] = lrun[r]*corr + rs;
      mrun[r] = mnew;
      #pragma unroll
      for (int j = 0; j < 4; ++j) o[j][r] = o[j][r]*corr;
      p0[r] = e0; p1[r] = e1;
    }

    // P (C-layout) -> LDS -> A-fragment layout (DS ops are in-order per wave)
    __bf16* pl = plds[wave];
    #pragma unroll
    for (int r = 0; r < 8; ++r) {
      pl[(r + 8*half)*32 + mrow]      = (__bf16)p0[r];
      pl[(r + 8*half)*32 + 16 + mrow] = (__bf16)p1[r];
    }
    __asm__ volatile("s_wait_dscnt 0" ::: "memory");
    const __bf16* prow = pl + mrow*32;
    const v16bf pa = cat8(*(const v8bf*)(prow + 16*half), *(const v8bf*)(prow + 16*half + 8));

    // O += P @ V  (Vt rows give contiguous B-fragments)
    #pragma unroll
    for (int j = 0; j < 4; ++j) {
      const __bf16* vp = Vt + ((size_t)bh*DK + 16*j + mrow)*SEQL + kbase + 16*half;
      o[j] = wmma_bf16(pa, *(const v16bf*)vp, o[j]);
    }
  }

  #pragma unroll
  for (int r = 0; r < 8; ++r) {
    const float inv = 1.f / lrun[r];
    const size_t row = (size_t)(b*SEQL + qbase + r + 8*half);
    #pragma unroll
    for (int j = 0; j < 4; ++j)
      O[row*DMODEL + h*DK + 16*j + mrow] = (__bf16)(o[j][r] * inv);
  }
}

// ---------------- LayerNorm(x + res), writes f32 and/or bf16 ----------------
__global__ __launch_bounds__(256)
void k_layernorm(const float* __restrict__ x, const float* __restrict__ res,
                 const float* __restrict__ gamma, const float* __restrict__ beta,
                 float* __restrict__ yf, __bf16* __restrict__ yb)
{
  __shared__ float sred[16];
  const int tid = threadIdx.x;
  const size_t row = blockIdx.x;
  const float* xp = x   + row*DMODEL;
  const float* rp = res + row*DMODEL;
  float v0 = xp[tid] + rp[tid];
  float v1 = xp[tid+256] + rp[tid+256];
  float s = v0 + v1, ss = v0*v0 + v1*v1;
  #pragma unroll
  for (int m = 1; m < 32; m <<= 1) { s += __shfl_xor(s, m, 32); ss += __shfl_xor(ss, m, 32); }
  if ((tid & 31) == 0) { sred[tid>>5] = s; sred[8 + (tid>>5)] = ss; }
  __syncthreads();
  float ts = 0.f, tss = 0.f;
  #pragma unroll
  for (int i = 0; i < 8; ++i) { ts += sred[i]; tss += sred[8+i]; }
  const float mean = ts * (1.f/DMODEL);
  const float var  = tss * (1.f/DMODEL) - mean*mean;
  const float rinv = rsqrtf(var + 1e-15f);
  const float y0 = gamma[tid]*(v0-mean)*rinv + beta[tid];
  const float y1 = gamma[tid+256]*(v1-mean)*rinv + beta[tid+256];
  if (yf) { yf[row*DMODEL+tid] = y0; yf[row*DMODEL+tid+256] = y1; }
  if (yb) { yb[row*DMODEL+tid] = (__bf16)y0; yb[row*DMODEL+tid+256] = (__bf16)y1; }
}

// ---------------- in-place row softmax (vocab) ----------------
__global__ __launch_bounds__(256)
void k_softmax_rows(float* __restrict__ x, int N)
{
  __shared__ float sred[8];
  float* row = x + (size_t)blockIdx.x*N;
  const int tid = threadIdx.x;
  float mx = -INFINITY;
  for (int i = tid; i < N; i += 256) mx = fmaxf(mx, row[i]);
  #pragma unroll
  for (int m = 1; m < 32; m <<= 1) mx = fmaxf(mx, __shfl_xor(mx, m, 32));
  if ((tid & 31) == 0) sred[tid>>5] = mx;
  __syncthreads();
  float rmx = sred[0];
  #pragma unroll
  for (int i = 1; i < 8; ++i) rmx = fmaxf(rmx, sred[i]);
  __syncthreads();
  float sum = 0.f;
  for (int i = tid; i < N; i += 256) { float e = __expf(row[i]-rmx); row[i] = e; sum += e; }
  #pragma unroll
  for (int m = 1; m < 32; m <<= 1) sum += __shfl_xor(sum, m, 32);
  if ((tid & 31) == 0) sred[tid>>5] = sum;
  __syncthreads();
  float tsum = 0.f;
  #pragma unroll
  for (int i = 0; i < 8; ++i) tsum += sred[i];
  const float inv = 1.f / tsum;
  for (int i = tid; i < N; i += 256) row[i] *= inv;
}

// =========================== host orchestration ===========================
extern "C" void kernel_launch(void* const* d_in, const int* in_sizes, int n_in,
                              void* d_out, int out_size, void* d_ws, size_t ws_size,
                              hipStream_t stream)
{
  (void)in_sizes; (void)n_in; (void)out_size; (void)ws_size;
  const float* src = (const float*)d_in[0];
  const float* trg = (const float*)d_in[1];
  auto in_f = [&](int i) { return (const float*)d_in[i]; };

  // -------- workspace carve-out --------
  char* ws = (char*)d_ws;
  size_t off = 0;
  auto carve = [&](size_t bytes) { void* p = ws + off; off += (bytes + 255) & ~(size_t)255; return p; };

  __bf16* wb[16];
  for (int j = 0; j < 16; ++j) wb[j] = (__bf16*)carve((size_t)DMODEL*DMODEL*2);
  __bf16* wob  = (__bf16*)carve((size_t)NVOCAB*DMODEL*2);
  __bf16* srcb = (__bf16*)carve((size_t)NROWS*DMODEL*2);
  __bf16* trgb = (__bf16*)carve((size_t)NROWS*DMODEL*2);
  __bf16* n1eb = (__bf16*)carve((size_t)NROWS*DMODEL*2);
  __bf16* n2eb = (__bf16*)carve((size_t)NROWS*DMODEL*2);
  __bf16* n2db = (__bf16*)carve((size_t)NROWS*DMODEL*2);
  __bf16* n3db = (__bf16*)carve((size_t)NROWS*DMODEL*2);
  __bf16* ffhb = (__bf16*)carve((size_t)NROWS*DMODEL*2);
  __bf16* acatb= (__bf16*)carve((size_t)NROWS*DMODEL*2);
  float*  Qf   = (float*)carve((size_t)NROWS*DMODEL*4);
  float*  Kf   = (float*)carve((size_t)NROWS*DMODEL*4);
  float*  Vf   = (float*)carve((size_t)NROWS*DMODEL*4);
  __bf16* qhb  = (__bf16*)carve((size_t)NROWS*DMODEL*2);
  __bf16* khb  = (__bf16*)carve((size_t)NROWS*DMODEL*2);
  __bf16* vhb  = (__bf16*)carve((size_t)NROWS*DMODEL*2);
  float*  X    = (float*)carve((size_t)NROWS*DMODEL*4);
  float*  n1e  = (float*)carve((size_t)NROWS*DMODEL*4);
  float*  n1d  = (float*)carve((size_t)NROWS*DMODEL*4);
  float*  n2d  = (float*)carve((size_t)NROWS*DMODEL*4);

  // -------- one-time (per call) bf16 conversions --------
  // weight input indices: enc{2,4,6,8} masked{10,12,14,16} cross{18,20,22,24} ff1{26,28} ff2{30,32}
  const int widx[16] = {2,4,6,8, 10,12,14,16, 18,20,22,24, 26,28, 30,32};
  for (int j = 0; j < 16; ++j)
    k_cvt_bf16<<<(DMODEL*DMODEL)/256, 256, 0, stream>>>(in_f(widx[j]), wb[j], DMODEL*DMODEL);
  k_cvt_bf16<<<(NVOCAB*DMODEL)/256, 256, 0, stream>>>(in_f(44), wob, NVOCAB*DMODEL);
  k_cvt_bf16<<<(NROWS*DMODEL)/256, 256, 0, stream>>>(src, srcb, NROWS*DMODEL);
  k_cvt_bf16<<<(NROWS*DMODEL)/256, 256, 0, stream>>>(trg, trgb, NROWS*DMODEL);

  auto gemm = [&](const __bf16* A, const __bf16* Wm, const float* bias,
                  float* Cf, __bf16* Cb, int M, int N, int K, int relu) {
    dim3 grid(M/128, N/64);
    k_gemm_bf16<<<grid, 256, 0, stream>>>(A, Wm, bias, Cf, Cb, M, N, K, relu);
  };

  auto mha = [&](const __bf16* xq, const __bf16* xkv, int base,
                 const __bf16* wq, const __bf16* wk, const __bf16* wv, const __bf16* wo,
                 int causal, float* Xout) {
    gemm(xq,  wq, in_f(base+1), Qf, nullptr, NROWS, DMODEL, DMODEL, 0);
    gemm(xkv, wk, in_f(base+3), Kf, nullptr, NROWS, DMODEL, DMODEL, 0);
    gemm(xkv, wv, in_f(base+5), Vf, nullptr, NROWS, DMODEL, DMODEL, 0);
    const int npk = NROWS*DMODEL;
    k_pack_heads<<<npk/256, 256, 0, stream>>>(Qf, qhb, 0);
    k_pack_heads<<<npk/256, 256, 0, stream>>>(Kf, khb, 0);
    k_pack_heads<<<npk/256, 256, 0, stream>>>(Vf, vhb, 1);
    k_flash_attn<<<dim3(NBATCH*NHEADS, SEQL/128), 256, 0, stream>>>(qhb, khb, vhb, acatb, causal);
    gemm(acatb, wo, in_f(base+7), Xout, nullptr, NROWS, DMODEL, DMODEL, 0);
  };

  // -------- encoder --------
  mha(srcb, srcb, 2, wb[0], wb[1], wb[2], wb[3], 0, X);
  k_layernorm<<<NROWS, 256, 0, stream>>>(X, src, in_f(34), in_f(35), n1e, n1eb);   // norm1_e
  gemm(n1eb, wb[12], in_f(27), nullptr, ffhb, NROWS, DMODEL, DMODEL, 1);           // ff1.fc1 + relu
  gemm(ffhb, wb[13], in_f(29), X, nullptr, NROWS, DMODEL, DMODEL, 0);              // ff1.fc2
  k_layernorm<<<NROWS, 256, 0, stream>>>(X, n1e, in_f(36), in_f(37), nullptr, n2eb); // norm2_e

  // -------- decoder --------
  mha(trgb, trgb, 10, wb[4], wb[5], wb[6], wb[7], 1, X);                           // masked attn
  k_layernorm<<<NROWS, 256, 0, stream>>>(X, trg, in_f(38), in_f(39), n1d, nullptr); // norm1_d
  mha(n1eb, n2eb, 18, wb[8], wb[9], wb[10], wb[11], 0, X);                         // cross attn
  k_layernorm<<<NROWS, 256, 0, stream>>>(X, n1d, in_f(40), in_f(41), n2d, n2db);   // norm2_d
  gemm(n2db, wb[14], in_f(31), nullptr, ffhb, NROWS, DMODEL, DMODEL, 1);           // ff2.fc1 + relu
  gemm(ffhb, wb[15], in_f(33), X, nullptr, NROWS, DMODEL, DMODEL, 0);              // ff2.fc2
  k_layernorm<<<NROWS, 256, 0, stream>>>(X, n2d, in_f(42), in_f(43), nullptr, n3db); // norm3_d

  // -------- vocab projection + softmax (in place on d_out) --------
  gemm(n3db, wob, in_f(45), (float*)d_out, nullptr, NROWS, NVOCAB, DMODEL, 0);
  k_softmax_rows<<<NROWS, 256, 0, stream>>>((float*)d_out, NVOCAB);
}
